// ThymeLM_71554155151769
// MI455X (gfx1250) — matrix operand order repaired and checked by reading, not compiled, not run on hardware
//
#include <hip/hip_runtime.h>
#include <math.h>

typedef __attribute__((ext_vector_type(16))) __bf16 v16bf;
typedef __attribute__((ext_vector_type(8)))  __bf16 v8bf;
typedef __attribute__((ext_vector_type(8)))  float  v8f;

#define V_SZ 32000
#define E_SZ 256
#define B_SZ 8
#define S_SZ 512
#define NA 12
#define NC 24
#define NS 576
#define NTOK (B_SZ * S_SZ)   // 4096
#define EPS 1e-5f
#define DECAY 0.9f

#define TM 64      // output rows per block (4 M-tiles per wave)
#define TN 128     // output cols per block (8 waves x 16)
#define PITCH 40   // LDS pitch (elements): 80B rows -> 16B-aligned reads, 20-dword bank spread

// ---------------------------------------------------------------- utilities
__device__ inline float wave_sum32(float v) {
  #pragma unroll
  for (int m = 16; m >= 1; m >>= 1) v += __shfl_xor(v, m, 32);
  return v;
}

// ---------------------------------------------------------------- fp32 -> bf16 convert
__global__ void thyme_cvt_bf16(const float* __restrict__ src,
                               __bf16* __restrict__ dst, int n) {
  int i = blockIdx.x * blockDim.x + threadIdx.x;
  if (i < n) dst[i] = (__bf16)src[i];
}

// ---------------------------------------------------------------- token branch:
// comps[tok,0:24] = tanh(LN(tanh(LN(emb@ax_w+ax_b)) @ comp_w + comp_b))
// one wave per token; 8 waves per block
__global__ void thyme_comps(const int* __restrict__ ids,
                            const float* __restrict__ embed,
                            const float* __restrict__ ax_w, const float* __restrict__ ax_b,
                            const float* __restrict__ ax_g, const float* __restrict__ ax_be,
                            const float* __restrict__ comp_w, const float* __restrict__ comp_b,
                            const float* __restrict__ comp_g, const float* __restrict__ comp_be,
                            float* __restrict__ comps) {
  const int lane = threadIdx.x & 31;
  const int wave = threadIdx.x >> 5;
  const int tok  = blockIdx.x * 8 + wave;
  if (tok >= NTOK) return;

  const float* erow = embed + (size_t)ids[tok] * E_SZ;

  float part[NA];
  #pragma unroll
  for (int a = 0; a < NA; ++a) part[a] = 0.f;
  #pragma unroll
  for (int r = 0; r < E_SZ / 32; ++r) {
    const int e = lane + 32 * r;
    const float ev = erow[e];
    #pragma unroll
    for (int a = 0; a < NA; ++a) part[a] += ev * ax_w[e * NA + a];
  }
  float ax[NA];
  #pragma unroll
  for (int a = 0; a < NA; ++a) ax[a] = wave_sum32(part[a]) + ax_b[a];

  // LN over 12 (population var) + tanh, redundantly on all lanes
  float m = 0.f, s2 = 0.f;
  #pragma unroll
  for (int a = 0; a < NA; ++a) { m += ax[a]; s2 += ax[a] * ax[a]; }
  m /= (float)NA;
  const float rs_a = rsqrtf(s2 / (float)NA - m * m + EPS);
  #pragma unroll
  for (int a = 0; a < NA; ++a)
    ax[a] = tanhf((ax[a] - m) * rs_a * ax_g[a] + ax_be[a]);

  // composites: lane c (<24) owns one output
  float myc = 0.f;
  if (lane < NC) {
    #pragma unroll
    for (int a = 0; a < NA; ++a) myc += ax[a] * comp_w[a * NC + lane];
    myc += comp_b[lane];
  }
  const float s1c = wave_sum32(lane < NC ? myc : 0.f);
  const float s2c = wave_sum32(lane < NC ? myc * myc : 0.f);
  const float mc  = s1c / (float)NC;
  const float rsc = rsqrtf(s2c / (float)NC - mc * mc + EPS);
  if (lane < NC) {
    const float y = (myc - mc) * rsc * comp_g[lane] + comp_be[lane];
    comps[(size_t)tok * NC + lane] = tanhf(y);
  }
}

// ---------------------------------------------------------------- state prefix scan
__global__ void thyme_scan(const float* __restrict__ comps,
                           const float* __restrict__ mix,
                           __bf16* __restrict__ states_bf,
                           float* __restrict__ final_state) {
  const int tid = blockIdx.x * blockDim.x + threadIdx.x;
  if (tid >= B_SZ * NS) return;
  const int b   = tid / NS;
  const int rem = tid % NS;
  const int i   = rem / NC;
  const int j   = rem % NC;
  const float sig = 1.0f / (1.0f + expf(-mix[0]));
  float st = 0.f;
  for (int t = 0; t < S_SZ; ++t) {
    const size_t tok = (size_t)b * S_SZ + t;
    const float ci = comps[tok * NC + i];
    const float cj = comps[tok * NC + j];
    st = DECAY * st + sig * ci * cj;
    states_bf[tok * NS + rem] = (__bf16)st;
  }
  final_state[tid] = st;
}

// ---------------------------------------------------------------- bf16 WMMA GEMM
// C[M,N] = A[M,K] * B[K,N] + bias; optional exact GELU; f32 or bf16 output.
// block = 256 threads (8 waves). Block tile: 64 rows x 128 cols.
// Both operand tiles staged via LDS:
//   A tile [64 x 32] row-major (pitch 40): 1 global b128 + 1 ds_store_b128 per thread.
//   B tile [32 x 128] transposed to [n][k] (pitch 40): 2 global b128 + 16 b16 stores.
// Fragments: all ds_load_b128 (16B-aligned); 4 back-to-back WMMAs per wave per K-step.
__global__ void __launch_bounds__(256)
thyme_gemm_bf16(const __bf16* __restrict__ A, const __bf16* __restrict__ Bm,
                const float* __restrict__ bias,
                float* __restrict__ Cf, __bf16* __restrict__ Cb,
                int M, int N, int K, int do_gelu) {
  __shared__ __bf16 ldsA[TM * PITCH];    // 5  KB, [m][k]
  __shared__ __bf16 ldsB[TN * PITCH];    // 10 KB, [n][k]

  const int tid  = threadIdx.x;
  const int lane = tid & 31;
  const int wave = tid >> 5;
  const int m0 = blockIdx.y * TM;
  const int n0 = blockIdx.x * TN;
  const int mr  = lane & 15;
  const int grp = lane >> 4;
  const int nw  = n0 + wave * 16 + mr;   // this lane's output column

  // B staging: thread -> (k row, 8-wide n chunk); 2 passes cover 32x128
  const int sk = tid >> 4;               // 0..15
  const int sn = (tid & 15) * 8;         // 0..120
  // A staging: thread -> (m row, 8-wide k chunk); 1 pass covers 64x32
  const int ar = tid >> 2;               // 0..63
  const int ak = (tid & 3) * 8;          // 0,8,16,24

  v8f acc[4] = {{}, {}, {}, {}};

  for (int k0 = 0; k0 < K; k0 += 32) {
    __syncthreads();
    if (k0 + 32 < K) {  // prefetch next tiles (global_prefetch_b8)
      __builtin_prefetch(Bm + (size_t)(k0 + 32 + sk) * N + n0 + sn, 0, 0);
      __builtin_prefetch(A + (size_t)(m0 + ar) * K + k0 + 32 + ak, 0, 0);
    }
    // ---- stage A [64 x 32] row-major
    {
      const v8bf av = *(const v8bf*)(A + (size_t)(m0 + ar) * K + k0 + ak);
      *(v8bf*)&ldsA[ar * PITCH + ak] = av;
    }
    // ---- stage B [32 x 128] transposed -> [n][k]
    #pragma unroll
    for (int pass = 0; pass < 2; ++pass) {
      const int k = sk + pass * 16;
      const v8bf src = *(const v8bf*)(Bm + (size_t)(k0 + k) * N + n0 + sn);
      #pragma unroll
      for (int i = 0; i < 8; ++i) ldsB[(sn + i) * PITCH + k] = src[i];
    }
    __syncthreads();

    // ---- B fragment: lane = column nw, k = grp*8 .. +7, 16+grp*8 .. +7
    const __bf16* bp = &ldsB[(wave * 16 + mr) * PITCH + grp * 8];
    const v8bf blo = *(const v8bf*)bp;
    const v8bf bhi = *(const v8bf*)(bp + 16);
    v16bf bfrag;
    #pragma unroll
    for (int q = 0; q < 8; ++q) { bfrag[q] = blo[q]; bfrag[8 + q] = bhi[q]; }

    // ---- A fragments for all 4 M-tiles, then 4 grouped WMMAs
    v16bf afrag[4];
    #pragma unroll
    for (int mt = 0; mt < 4; ++mt) {
      const __bf16* ap = &ldsA[(mt * 16 + mr) * PITCH + grp * 8];
      const v8bf alo = *(const v8bf*)ap;          // K = grp*8    .. +7
      const v8bf ahi = *(const v8bf*)(ap + 16);   // K = 16+grp*8 .. +7
      #pragma unroll
      for (int q = 0; q < 8; ++q) { afrag[mt][q] = alo[q]; afrag[mt][8 + q] = ahi[q]; }
    }
    #pragma unroll
    for (int mt = 0; mt < 4; ++mt)
      acc[mt] = __builtin_amdgcn_wmma_f32_16x16x32_bf16(
          false, afrag[mt], false, bfrag, (short)0, acc[mt], false, false);
  }

  // epilogue: acc[mt][r] -> (m0 + mt*16 + grp*8 + r, nw)
  const float bv = bias[nw];
  #pragma unroll
  for (int mt = 0; mt < 4; ++mt) {
    #pragma unroll
    for (int r = 0; r < 8; ++r) {
      const int mm = m0 + mt * 16 + grp * 8 + r;
      float v = acc[mt][r] + bv;
      if (do_gelu) v = 0.5f * v * (1.0f + erff(v * 0.70710678118f));
      if (Cf) Cf[(size_t)mm * N + nw] = v;
      else    Cb[(size_t)mm * N + nw] = (__bf16)v;
    }
  }
}

// ---------------------------------------------------------------- launcher
extern "C" void kernel_launch(void* const* d_in, const int* in_sizes, int n_in,
                              void* d_out, int out_size, void* d_ws, size_t ws_size,
                              hipStream_t stream) {
  const int*   ids     = (const int*)  d_in[0];
  const float* embed   = (const float*)d_in[1];
  const float* ax_w    = (const float*)d_in[2];
  const float* ax_b    = (const float*)d_in[3];
  const float* ax_g    = (const float*)d_in[4];
  const float* ax_be   = (const float*)d_in[5];
  const float* comp_w  = (const float*)d_in[6];
  const float* comp_b  = (const float*)d_in[7];
  const float* comp_g  = (const float*)d_in[8];
  const float* comp_be = (const float*)d_in[9];
  const float* mix     = (const float*)d_in[10];
  const float* w1      = (const float*)d_in[11];
  const float* b1      = (const float*)d_in[12];
  const float* w2      = (const float*)d_in[13];
  const float* b2      = (const float*)d_in[14];
  float* out = (float*)d_out;                 // [B,S,V] logits ++ [B,24,24] state

  char* p = (char*)d_ws;
  float*  comps    = (float*)p;  p += (size_t)NTOK * NC * 4;
  __bf16* statesbf = (__bf16*)p; p += (size_t)NTOK * NS * 2;
  __bf16* Hbf      = (__bf16*)p; p += (size_t)NTOK * 512 * 2;
  __bf16* w1bf     = (__bf16*)p; p += (size_t)NS * 512 * 2;
  __bf16* w2bf     = (__bf16*)p; p += (size_t)512 * V_SZ * 2;

  thyme_cvt_bf16<<<(NS * 512 + 255) / 256, 256, 0, stream>>>(w1, w1bf, NS * 512);
  thyme_cvt_bf16<<<(512 * V_SZ + 255) / 256, 256, 0, stream>>>(w2, w2bf, 512 * V_SZ);

  thyme_comps<<<NTOK / 8, 256, 0, stream>>>(ids, embed, ax_w, ax_b, ax_g, ax_be,
                                            comp_w, comp_b, comp_g, comp_be, comps);

  thyme_scan<<<(B_SZ * NS + 255) / 256, 256, 0, stream>>>(
      comps, mix, statesbf, out + (size_t)NTOK * V_SZ);

  // GEMM1: H = gelu(states @ W1 + b1) -> bf16 [4096,512]
  thyme_gemm_bf16<<<dim3(512 / TN, NTOK / TM), 256, 0, stream>>>(
      statesbf, w1bf, b1, nullptr, Hbf, NTOK, 512, NS, /*gelu=*/1);

  // GEMM2: logits = H @ W2 + b2 -> f32 [4096,32000] (== [B,S,V])
  thyme_gemm_bf16<<<dim3(V_SZ / TN, NTOK / TM), 256, 0, stream>>>(
      Hbf, w2bf, b2, out, nullptr, NTOK, V_SZ, 512, /*gelu=*/0);
}
